// GCN_42898133353087
// MI455X (gfx1250) — matrix operand order repaired and checked by reading, not compiled, not run on hardware
//
#include <hip/hip_runtime.h>

// ---------------------------------------------------------------------------
// GCN forward, MI455X (gfx1250, wave32).
// Deep-layer GEMMs run on the matrix pipe via v_wmma_f32_16x16x32_f16 using
// split-precision f16x3 (hi/lo decomposition, f32 accumulate) for ~fp32
// accuracy. Edge aggregation is f32 atomic scatter-add (bandwidth-bound part).
// ---------------------------------------------------------------------------

typedef __attribute__((ext_vector_type(16))) _Float16 v16h;
typedef __attribute__((ext_vector_type(8)))  _Float16 v8h;
typedef __attribute__((ext_vector_type(8)))  float    v8f;

#define HDIM 256

__device__ __forceinline__ float leaky01(float v) { return v > 0.0f ? v : 0.01f * v; }

// ---------------- small utility kernels ----------------

__global__ void k_fill(float* __restrict__ p, float v, int n) {
    int i = blockIdx.x * blockDim.x + threadIdx.x;
    if (i < n) p[i] = v;
}

__global__ void k_deg_edges(const int* __restrict__ dst, float* __restrict__ deg, int E) {
    int e = blockIdx.x * blockDim.x + threadIdx.x;
    if (e < E) atomicAdd(&deg[dst[e]], 1.0f);
}

__global__ void k_rsqrt_inplace(float* __restrict__ p, int n) {
    int i = blockIdx.x * blockDim.x + threadIdx.x;
    if (i < n) p[i] = rsqrtf(p[i]);
}

__global__ void k_gather_x(const float* __restrict__ sst, const int* __restrict__ mask,
                           float* __restrict__ x0, int N, int F, int total) {
    int i = blockIdx.x * blockDim.x + threadIdx.x;
    if (i >= total) return;
    int b = i / N;
    int j = i - b * N;
    x0[i] = sst[(size_t)b * F + mask[j]];
}

// ---------------- layer 1 (rank-1 trick: agg = s[i] * W1[j]) ----------------

__global__ void k_s_init(const float* __restrict__ dis, const float* __restrict__ x0,
                         float* __restrict__ s, int n) {
    int i = blockIdx.x * blockDim.x + threadIdx.x;
    if (i < n) s[i] = dis[i] * dis[i] * x0[i];   // self-loop: (1/deg) * x
}

__global__ void k_s_edge(const int* __restrict__ src, const int* __restrict__ dst,
                         const float* __restrict__ dis, const float* __restrict__ x0,
                         float* __restrict__ s, int E) {
    int e = blockIdx.x * blockDim.x + threadIdx.x;
    if (e >= E) return;
    int a = src[e], d = dst[e];
    atomicAdd(&s[d], dis[a] * dis[d] * x0[a]);
}

__global__ void k_layer1(const float* __restrict__ x0, const float* __restrict__ s,
                         const float* __restrict__ W1, const float* __restrict__ b1,
                         float* __restrict__ h, int total) {
    int t = blockIdx.x * blockDim.x + threadIdx.x;
    if (t >= total) return;
    int i = t >> 8;
    int j = t & (HDIM - 1);
    h[t] = x0[i] + leaky01(s[i] * W1[j] + b1[j]);
}

// ---------------- f32 -> split f16 (hi + lo) activation convert ----------------

__global__ void k_split_f16(const float* __restrict__ h, _Float16* __restrict__ hhi,
                            _Float16* __restrict__ hlo, int total) {
    int t = blockIdx.x * blockDim.x + threadIdx.x;
    if (t >= total) return;
    float v  = h[t];
    _Float16 hi = (_Float16)v;
    hhi[t] = hi;
    hlo[t] = (_Float16)(v - (float)hi);
}

// ---------------- pack W (f32 row-major [K=256][N=256]) into WMMA B-fragment order ------
// element e of the v16h per lane maps to K = ks*32 + (e<8 ? 8*(lane>>4)+e : 16+8*(lane>>4)+(e-8)),  N = nt*16 + (lane&15)
// Writes split hi/lo fragment streams.

__global__ void k_packW(const float* __restrict__ Ws, _Float16* __restrict__ Bhi,
                        _Float16* __restrict__ Blo, int total) {
    int t = blockIdx.x * blockDim.x + threadIdx.x;   // t = ((l*16+nt)*8+ks)*32+lane
    if (t >= total) return;
    int lane = t & 31;
    int ks   = (t >> 5) & 7;
    int nt   = (t >> 8) & 15;
    int l    = t >> 12;
    int half = lane >> 4;
    int n    = nt * 16 + (lane & 15);
    const float* W = Ws + (size_t)l * HDIM * HDIM;
    _Float16* ohi = Bhi + (size_t)t * 16;
    _Float16* olo = Blo + (size_t)t * 16;
#pragma unroll
    for (int e = 0; e < 16; ++e) {
        int k = ks * 32 + (e < 8 ? 8 * half + e : 16 + 8 * half + (e - 8));
        float v  = W[(size_t)k * HDIM + n];
        _Float16 hi = (_Float16)v;
        ohi[e] = hi;
        olo[e] = (_Float16)(v - (float)hi);
    }
}

// ---------------- WMMA GEMM: HW[NT,256] = h[NT,256] @ W_l[256,256], f16x3 split ---------
// grid = (NT/16, 2), block = 256 (8 waves). Each wave owns one 16x16 output tile.
// Per K-step: acc += Ahi*Bhi + Alo*Bhi + Ahi*Blo   (Alo*Blo term ~2^-22, dropped)

__global__ void __launch_bounds__(256) k_gemm_wmma(const _Float16* __restrict__ hhi,
                                                   const _Float16* __restrict__ hlo,
                                                   const _Float16* __restrict__ Bhi,
                                                   const _Float16* __restrict__ Blo,
                                                   float* __restrict__ HW, int layer) {
    const int lane = threadIdx.x & 31;
    const int wave = threadIdx.x >> 5;
    const int mt   = blockIdx.x;
    const int nt   = blockIdx.y * 8 + wave;
    const int half = lane >> 4;
    const int mr   = lane & 15;

    // A fragment sources: row (mt*16+mr), K chunks split across lane halves per ISA layout
    const size_t rowoff = ((size_t)mt * 16 + mr) * HDIM + 8 * half;
    const _Float16* arowh = hhi + rowoff;
    const _Float16* arowl = hlo + rowoff;
    const size_t boff = (((size_t)layer * 16 + nt) * 8 * 32 + lane) * 16;
    const _Float16* bph = Bhi + boff;
    const _Float16* bpl = Blo + boff;

    v8f acc = {};
#pragma unroll
    for (int ks = 0; ks < 8; ++ks) {
        v8h ah0 = *(const v8h*)(arowh + ks * 32);        // K = kb + 8*half + 0..7
        v8h ah1 = *(const v8h*)(arowh + ks * 32 + 16);   // K = kb + 16 + 8*half + 0..7
        v8h al0 = *(const v8h*)(arowl + ks * 32);
        v8h al1 = *(const v8h*)(arowl + ks * 32 + 16);
        v16h ahi = __builtin_shufflevector(ah0, ah1, 0, 1, 2, 3, 4, 5, 6, 7,
                                                     8, 9, 10, 11, 12, 13, 14, 15);
        v16h alo = __builtin_shufflevector(al0, al1, 0, 1, 2, 3, 4, 5, 6, 7,
                                                     8, 9, 10, 11, 12, 13, 14, 15);
        v16h bhi = *(const v16h*)(bph + (size_t)ks * 32 * 16);
        v16h blo = *(const v16h*)(bpl + (size_t)ks * 32 * 16);
        acc = __builtin_amdgcn_wmma_f32_16x16x32_f16(false, ahi, false, bhi,
                                                     (short)0, acc, false, false);
        acc = __builtin_amdgcn_wmma_f32_16x16x32_f16(false, alo, false, bhi,
                                                     (short)0, acc, false, false);
        acc = __builtin_amdgcn_wmma_f32_16x16x32_f16(false, ahi, false, blo,
                                                     (short)0, acc, false, false);
    }

    // D layout: VGPR r holds row (r + 8*half), column = lane&15 of the 16x16 tile
    float* out = HW + ((size_t)mt * 16 + 8 * half) * HDIM + nt * 16 + (lane & 15);
#pragma unroll
    for (int r = 0; r < 8; ++r) out[(size_t)r * HDIM] = acc[r];
}

// ---------------- edge aggregation: agg[d,:] += norm_e * HW[s,:]; self-loop in init -------

__global__ void k_agg_init(const float* __restrict__ dis, const float* __restrict__ HW,
                           float* __restrict__ agg, int total) {
    int t = blockIdx.x * blockDim.x + threadIdx.x;
    if (t >= total) return;
    int i = t >> 8;
    agg[t] = dis[i] * dis[i] * HW[t];
}

__global__ void k_agg_edge(const int* __restrict__ src, const int* __restrict__ dst,
                           const float* __restrict__ dis, const float* __restrict__ HW,
                           float* __restrict__ agg, int E) {
    long long t = (long long)blockIdx.x * blockDim.x + threadIdx.x;   // E*64 threads
    int e = (int)(t >> 6);
    if (e >= E) return;
    int c = (int)(t & 63) << 2;
    int s = src[e], d = dst[e];
    float nm = dis[s] * dis[d];
    const float4 v = *(const float4*)(HW + (size_t)s * HDIM + c);
    // warm the next source row chunk (emits global_prefetch_b8)
    __builtin_prefetch(HW + (size_t)s * HDIM + ((c + 64) & (HDIM - 1)), 0, 0);
    float* a = agg + (size_t)d * HDIM + c;
    atomicAdd(a + 0, nm * v.x);
    atomicAdd(a + 1, nm * v.y);
    atomicAdd(a + 2, nm * v.z);
    atomicAdd(a + 3, nm * v.w);
}

__global__ void k_residual(const float* __restrict__ agg, const float* __restrict__ b,
                           float* __restrict__ h, int total) {
    int t = blockIdx.x * blockDim.x + threadIdx.x;
    if (t >= total) return;
    int j = t & (HDIM - 1);
    h[t] = h[t] + leaky01(agg[t] + b[j]);
}

// ---------------- pooling + head ----------------

__global__ void k_pool(const float* __restrict__ h, float* __restrict__ pooled, int N) {
    int col = threadIdx.x;                       // 256 threads = one column each
    long long row0 = (long long)blockIdx.x * 128;
    int b = (int)(row0 / N);
    float sum = 0.0f;
#pragma unroll 4
    for (int r = 0; r < 128; ++r) sum += h[(row0 + r) * HDIM + col];
    atomicAdd(&pooled[b * HDIM + col], sum);
}

__global__ void k_head(const float* __restrict__ pooled, const float* __restrict__ Wh,
                       const float* __restrict__ bh, float* __restrict__ out, int N) {
    int b = blockIdx.x;
    int o = threadIdx.x;
    float inv = 1.0f / (float)N;
    float acc = bh[o];
    for (int k = 0; k < HDIM; ++k)
        acc += (pooled[b * HDIM + k] * inv) * Wh[(size_t)k * HDIM + o];
    out[b * HDIM + o] = acc;
}

// ---------------------------------------------------------------------------

extern "C" void kernel_launch(void* const* d_in, const int* in_sizes, int n_in,
                              void* d_out, int out_size, void* d_ws, size_t ws_size,
                              hipStream_t stream) {
    const float* sst  = (const float*)d_in[0];
    const int*   mask = (const int*)  d_in[1];
    const int*   eidx = (const int*)  d_in[2];
    const float* W1   = (const float*)d_in[3];
    const float* b1   = (const float*)d_in[4];
    const float* Ws   = (const float*)d_in[5];
    const float* bs   = (const float*)d_in[6];
    const float* Wh   = (const float*)d_in[7];
    const float* bh   = (const float*)d_in[8];

    const int N     = in_sizes[1];
    const int E     = in_sizes[2] / 2;
    const int B     = out_size / HDIM;
    const int F     = in_sizes[0] / B;
    const int DEPTH = in_sizes[6] / HDIM;
    const int NT    = B * N;
    const int TOT   = NT * HDIM;          // 33.5M elements

    const int* src = eidx;
    const int* dst = eidx + E;

    // -------- workspace layout (all offsets 256B aligned) --------
    char* w = (char*)d_ws;
    float*    dis    = (float*)w;    w += (size_t)NT * 4;             // deg -> dis in place
    float*    x0     = (float*)w;    w += (size_t)NT * 4;
    float*    svec   = (float*)w;    w += (size_t)NT * 4;
    float*    h      = (float*)w;    w += (size_t)TOT * 4;
    _Float16* hhi    = (_Float16*)w; w += (size_t)TOT * 2;
    _Float16* hlo    = (_Float16*)w; w += (size_t)TOT * 2;
    float*    HW     = (float*)w;    w += (size_t)TOT * 4;
    float*    agg    = (float*)w;    w += (size_t)TOT * 4;
    _Float16* Bhi    = (_Float16*)w; w += (size_t)DEPTH * HDIM * HDIM * 2;
    _Float16* Blo    = (_Float16*)w; w += (size_t)DEPTH * HDIM * HDIM * 2;
    float*    pooled = (float*)w;    w += (size_t)B * HDIM * 4;
    (void)ws_size; (void)n_in;

    const int TB = 256;
    dim3 blk(TB);

    // -------- degrees / norms (gcn_norm, cached once) --------
    k_fill<<<dim3((NT + TB - 1) / TB), blk, 0, stream>>>(dis, 1.0f, NT);  // self-loop +1
    k_deg_edges<<<dim3((E + TB - 1) / TB), blk, 0, stream>>>(dst, dis, E);
    k_rsqrt_inplace<<<dim3((NT + TB - 1) / TB), blk, 0, stream>>>(dis, NT);

    // -------- input gather --------
    k_gather_x<<<dim3((NT + TB - 1) / TB), blk, 0, stream>>>(sst, mask, x0, N, F, NT);

    // -------- layer 1 (rank-1) --------
    k_s_init<<<dim3((NT + TB - 1) / TB), blk, 0, stream>>>(dis, x0, svec, NT);
    k_s_edge<<<dim3((E + TB - 1) / TB), blk, 0, stream>>>(src, dst, dis, x0, svec, E);
    k_layer1<<<dim3((TOT + TB - 1) / TB), blk, 0, stream>>>(x0, svec, W1, b1, h, TOT);

    // -------- pack all deep-layer weights into WMMA fragment order (hi/lo split) --------
    {
        int packTot = DEPTH * 16 * 8 * 32;
        k_packW<<<dim3((packTot + TB - 1) / TB), blk, 0, stream>>>(Ws, Bhi, Blo, packTot);
    }

    // -------- deep layers: split-convert -> WMMA GEMM (f16x3) -> aggregate -> residual ----
    for (int l = 0; l < DEPTH; ++l) {
        k_split_f16<<<dim3((TOT + TB - 1) / TB), blk, 0, stream>>>(h, hhi, hlo, TOT);
        k_gemm_wmma<<<dim3(NT / 16, 2), blk, 0, stream>>>(hhi, hlo, Bhi, Blo, HW, l);
        k_agg_init<<<dim3((TOT + TB - 1) / TB), blk, 0, stream>>>(dis, HW, agg, TOT);
        {
            long long et = (long long)E * 64;
            k_agg_edge<<<dim3((unsigned)((et + TB - 1) / TB)), blk, 0, stream>>>(
                src, dst, dis, HW, agg, E);
        }
        k_residual<<<dim3((TOT + TB - 1) / TB), blk, 0, stream>>>(agg, bs + (size_t)l * HDIM, h, TOT);
    }

    // -------- mean pool + head --------
    k_fill<<<dim3((B * HDIM + TB - 1) / TB), blk, 0, stream>>>(pooled, 0.0f, B * HDIM);
    k_pool<<<dim3(NT / 128), blk, 0, stream>>>(h, pooled, N);
    k_head<<<dim3(B), blk, 0, stream>>>(pooled, Wh, bh, (float*)d_out, N);
}
